// CasualAttention_11252814315618
// MI455X (gfx1250) — matrix operand order repaired and checked by reading, not compiled, not run on hardware
//
#include <hip/hip_runtime.h>

// ---------------------------------------------------------------------------
// Causal multi-head attention for MI455X (gfx1250, wave32, WMMA bf16)
// B=4, T=2048, C=1024, H=16, hd=64
//
// Pipeline:
//   0a) cvt_x:         x f32[8192,1024] -> bf16 [8192,1024]
//   0b) transpose_cvt: w f32[K,N]       -> bf16 [N,K]   (w_qkv, w_out)
//   1) qkv_gemm:   xbf @ wqkvT -> q,k bf16 [B,H,T,64], v^T bf16 [B,H,64,T]
//       32x64 per wave, register double-buffered K loop (8 WMMA / step)
//   2) flash_attn: causal online-softmax, QK^T and PV via wmma bf16
//   3) out_proj:   attn(bf16) @ woutT(bf16) -> d_out f32
// ---------------------------------------------------------------------------

typedef __bf16 bf16;
typedef __attribute__((ext_vector_type(16))) __bf16 v16bf;
typedef __attribute__((ext_vector_type(8)))  __bf16 v8bf;
typedef __attribute__((ext_vector_type(4)))  __bf16 v4bf;
typedef __attribute__((ext_vector_type(8)))  float  v8f;

#define WMMA_BF16(a, b, c) \
  __builtin_amdgcn_wmma_f32_16x16x32_bf16(false, (a), false, (b), (short)0, (c), false, false)

static constexpr int Bb = 4, Tt = 2048, Cc = 1024, Hh = 16, HD = 64;
static constexpr int Mrows = Bb * Tt;   // 8192
static constexpr int Nqkv  = 3 * Cc;    // 3072

__device__ __forceinline__ v16bf cat8(v8bf lo, v8bf hi) {
  return __builtin_shufflevector(lo, hi, 0, 1, 2, 3, 4, 5, 6, 7,
                                 8, 9, 10, 11, 12, 13, 14, 15);
}

// A-fragment (16x32) from a bf16 row: contiguous K order,
// element e <-> K = half*16+e (matches the B fragment).
__device__ __forceinline__ v16bf load_a_bf16(const bf16* __restrict__ row,
                                             int kb, int half) {
  return cat8(*(const v8bf*)(row + kb + half * 16),
              *(const v8bf*)(row + kb + half * 16 + 8));
}

// B-fragment (32x16) from bf16 [N,K] (transposed weights): lane carries column
// n, element e <-> K = half*16 + e -> one contiguous 32B load per lane.
__device__ __forceinline__ v16bf load_b_bf16T(const bf16* __restrict__ WT,
                                              int ldk, int n, int kb, int half) {
  return *(const v16bf*)(WT + (size_t)n * ldk + kb + half * 16);
}

// ---------------------------------------------------------------------------
// Kernel 0a: x f32 -> bf16 elementwise (vectorized 4/thread).
// ---------------------------------------------------------------------------
__global__ __launch_bounds__(256) void cvt_x_kernel(
    const float* __restrict__ in, bf16* __restrict__ out) {
  const size_t i = ((size_t)blockIdx.x * 256 + threadIdx.x) * 4;
  const float4 f = *(const float4*)(in + i);
  v4bf o;
  o[0] = (bf16)f.x; o[1] = (bf16)f.y; o[2] = (bf16)f.z; o[3] = (bf16)f.w;
  *(v4bf*)(out + i) = o;
}

// ---------------------------------------------------------------------------
// Kernel 0b: LDS-tiled transpose + convert: f32 [K,N] -> bf16 [N,K].
// grid = (K/64, N/64), block = 256. Coalesced reads and writes.
// ---------------------------------------------------------------------------
__global__ __launch_bounds__(256) void transpose_cvt_kernel(
    const float* __restrict__ in, bf16* __restrict__ out, int K, int N) {
  __shared__ float tile[64][65];
  const int kt = blockIdx.x * 64;
  const int nt = blockIdx.y * 64;
#pragma unroll
  for (int i = threadIdx.x; i < 64 * 64; i += 256) {
    const int r = i >> 6, c = i & 63;                 // r: k, c: n
    tile[r][c] = in[(size_t)(kt + r) * N + nt + c];
  }
  __syncthreads();
#pragma unroll
  for (int i = threadIdx.x; i < 64 * 64; i += 256) {
    const int r = i >> 6, c = i & 63;                 // r: n, c: k
    out[(size_t)(nt + r) * K + kt + c] = (bf16)tile[c][r];
  }
}

// ---------------------------------------------------------------------------
// Kernel 1: QKV projection. Block = 256 (8 waves); block tile 256x64; each
// wave computes a 32x64 strip. Double-buffered fragments: next K-step's loads
// issue before the current step's 8 WMMAs so the matrix pipe never waits on
// a full loadcnt drain.
// ---------------------------------------------------------------------------
__global__ __launch_bounds__(256) void qkv_gemm_kernel(
    const bf16* __restrict__ xbf, const bf16* __restrict__ wqkvT,
    bf16* __restrict__ qws, bf16* __restrict__ kws, bf16* __restrict__ vws) {
  const int lane   = threadIdx.x & 31;
  const int wave   = threadIdx.x >> 5;
  const int lane15 = lane & 15;
  const int half   = lane >> 4;
  const int m0 = blockIdx.y * 256 + wave * 32;
  const int n0 = blockIdx.x * 64;

  const bf16* arow0 = xbf + (size_t)(m0 + lane15) * Cc;
  const bf16* arow1 = xbf + (size_t)(m0 + 16 + lane15) * Cc;
  v8f acc[2][4] = {};

  // Prologue fragments (K-step 0).
  v16bf a0 = load_a_bf16(arow0, 0, half);
  v16bf a1 = load_a_bf16(arow1, 0, half);
  v16bf bc[4];
#pragma unroll
  for (int j = 0; j < 4; ++j)
    bc[j] = load_b_bf16T(wqkvT, Cc, n0 + j * 16 + lane15, 0, half);

#pragma unroll 2
  for (int kb = 0; kb < Cc; kb += 32) {
    const int kn = (kb + 32 < Cc) ? (kb + 32) : kb;   // clamped (safe reload)
    __builtin_prefetch(arow0 + kn + 96, 0, 1);
    const v16bf na0 = load_a_bf16(arow0, kn, half);
    const v16bf na1 = load_a_bf16(arow1, kn, half);
    v16bf nb[4];
#pragma unroll
    for (int j = 0; j < 4; ++j)
      nb[j] = load_b_bf16T(wqkvT, Cc, n0 + j * 16 + lane15, kn, half);
#pragma unroll
    for (int j = 0; j < 4; ++j) {
      acc[0][j] = WMMA_BF16(a0, bc[j], acc[0][j]);
      acc[1][j] = WMMA_BF16(a1, bc[j], acc[1][j]);
    }
    a0 = na0; a1 = na1;
#pragma unroll
    for (int j = 0; j < 4; ++j) bc[j] = nb[j];
  }

  // Epilogue: per 16-col subtile the q/k/v section, head and d-base are
  // lane-uniform (16 | n boundaries), so select the destination once.
#pragma unroll
  for (int j = 0; j < 4; ++j) {
    const int nb16 = n0 + j * 16;
    const int sec = nb16 >> 10;           // 0=q 1=k 2=v (uniform)
    const int c0  = nb16 & (Cc - 1);
    const int h   = c0 >> 6;              // uniform
    const int d   = (c0 & 63) + lane15;
#pragma unroll
    for (int s = 0; s < 2; ++s) {
#pragma unroll
      for (int r = 0; r < 8; ++r) {
        const int gm = m0 + s * 16 + r + 8 * half;
        const int b  = gm >> 11;
        const int t  = gm & (Tt - 1);
        const size_t bh = (size_t)b * Hh + h;
        const bf16 v = (bf16)acc[s][j][r];
        if (sec == 0)      qws[(bh * Tt + t) * HD + d] = v;
        else if (sec == 1) kws[(bh * Tt + t) * HD + d] = v;
        else               vws[(bh * HD + d) * Tt + t] = v;  // V^T
      }
    }
  }
}

// ---------------------------------------------------------------------------
// Kernel 2: causal flash attention. 128 threads = 4 waves; each wave owns 16
// query rows of one (b,h). Wave-uniform causal loop -> EXEC all-ones at WMMA.
// V fragments load at the top of each K-block so they overlap the S WMMAs and
// the softmax VALU chain.
// ---------------------------------------------------------------------------
__global__ __launch_bounds__(128) void flash_attn_kernel(
    const bf16* __restrict__ Q, const bf16* __restrict__ K,
    const bf16* __restrict__ VT, bf16* __restrict__ att) {
  __shared__ __align__(16) bf16 plds[4][16][32];  // per-wave P tile (16x32)

  const int lane   = threadIdx.x & 31;
  const int wave   = threadIdx.x >> 5;
  const int lane15 = lane & 15;
  const int half   = lane >> 4;
  const int bh = blockIdx.y;
  const int b  = bh >> 4, h = bh & 15;
  const int q0 = blockIdx.x * 64 + wave * 16;

  const bf16* Qp = Q  + (size_t)bh * Tt * HD;
  const bf16* Kp = K  + (size_t)bh * Tt * HD;
  const bf16* Vp = VT + (size_t)bh * HD * Tt;

  // Q fragments: two k=32 A-frags covering d=0..31 and d=32..63.
  const bf16* qrow = Qp + (size_t)(q0 + lane15) * HD;
  const v16bf qa0 = cat8(*(const v8bf*)(qrow + half * 16),
                         *(const v8bf*)(qrow + half * 16 + 8));
  const v16bf qa1 = cat8(*(const v8bf*)(qrow + 32 + half * 16),
                         *(const v8bf*)(qrow + 32 + half * 16 + 8));

  v8f o[4] = {};                 // O accumulator, 16x64 f32
  float mrow[8], lrow[8];
#pragma unroll
  for (int r = 0; r < 8; ++r) { mrow[r] = -3.0e38f; lrow[r] = 0.f; }

  const float sc = 0.125f;       // 1/sqrt(64)
  const int kend = q0 + 15;      // last key any row of this wave may see

#pragma unroll 1
  for (int kb = 0; kb <= kend; kb += 32) {
    // ---- issue K and V fragment loads up front ----
    const bf16* k0p = Kp + (size_t)(kb + lane15) * HD;
    const bf16* k1p = Kp + (size_t)(kb + 16 + lane15) * HD;
    const v16bf kf00 = *(const v16bf*)(k0p + half * 16);
    const v16bf kf01 = *(const v16bf*)(k0p + 32 + half * 16);
    const v16bf kf10 = *(const v16bf*)(k1p + half * 16);
    const v16bf kf11 = *(const v16bf*)(k1p + 32 + half * 16);
    v16bf vf[4];
#pragma unroll
    for (int dt = 0; dt < 4; ++dt)
      vf[dt] = *(const v16bf*)(Vp + (size_t)(dt * 16 + lane15) * Tt +
                               kb + half * 16);

    // ---- S = Q K^T for 32 keys (two 16x16 tiles) ----
    v8f s0 = {}, s1 = {};
    s0 = WMMA_BF16(qa0, kf00, s0);
    s0 = WMMA_BF16(qa1, kf01, s0);
    s1 = WMMA_BF16(qa0, kf10, s1);
    s1 = WMMA_BF16(qa1, kf11, s1);

    // ---- online softmax (C-frag: lane=col, VGPR r = row r+8*half) ----
#pragma unroll
    for (int r = 0; r < 8; ++r) {
      const int gq = q0 + r + 8 * half;
      float v0 = s0[r] * sc; if (kb + lane15 > gq)      v0 = -3.0e38f;
      float v1 = s1[r] * sc; if (kb + 16 + lane15 > gq) v1 = -3.0e38f;
      float mx = fmaxf(v0, v1);
#pragma unroll
      for (int off = 1; off < 16; off <<= 1)
        mx = fmaxf(mx, __shfl_xor(mx, off, 32));
      const float mnew  = fmaxf(mrow[r], mx);
      const float alpha = __expf(mrow[r] - mnew);
      const float p0 = __expf(v0 - mnew);
      const float p1 = __expf(v1 - mnew);
      float rs = p0 + p1;
#pragma unroll
      for (int off = 1; off < 16; off <<= 1)
        rs += __shfl_xor(rs, off, 32);
      lrow[r] = lrow[r] * alpha + rs;
      mrow[r] = mnew;
#pragma unroll
      for (int dt = 0; dt < 4; ++dt) o[dt][r] *= alpha;
      plds[wave][r + 8 * half][lane15]      = (bf16)p0;
      plds[wave][r + 8 * half][16 + lane15] = (bf16)p1;
    }

    // C-frag -> A-frag re-layout through LDS (wave-private, lockstep DS order)
    __builtin_amdgcn_wave_barrier();
    asm volatile("" ::: "memory");
    const bf16* pr = &plds[wave][lane15][0];
    const v16bf pa = cat8(*(const v8bf*)(pr + half * 16),
                          *(const v8bf*)(pr + half * 16 + 8));
    __builtin_amdgcn_wave_barrier();
    asm volatile("" ::: "memory");

    // ---- O += P @ V (V fragments already in registers) ----
#pragma unroll
    for (int dt = 0; dt < 4; ++dt)
      o[dt] = WMMA_BF16(pa, vf[dt], o[dt]);
  }

  // ---- normalize and store attn (bf16 [B*T, C]) ----
#pragma unroll
  for (int r = 0; r < 8; ++r) {
    const float inv = 1.0f / lrow[r];
    const int t = q0 + r + 8 * half;
    const size_t base = ((size_t)b * Tt + t) * Cc + h * HD;
#pragma unroll
    for (int dt = 0; dt < 4; ++dt)
      att[base + dt * 16 + lane15] = (bf16)(o[dt][r] * inv);
  }
}

// ---------------------------------------------------------------------------
// Kernel 3: output projection, attn(bf16) @ woutT(bf16 [N,K]) -> f32.
// Same 32x64-per-wave double-buffered structure as kernel 1.
// ---------------------------------------------------------------------------
__global__ __launch_bounds__(256) void out_proj_kernel(
    const bf16* __restrict__ A, const bf16* __restrict__ WT,
    float* __restrict__ out) {
  const int lane   = threadIdx.x & 31;
  const int wave   = threadIdx.x >> 5;
  const int lane15 = lane & 15;
  const int half   = lane >> 4;
  const int m0 = blockIdx.y * 256 + wave * 32;
  const int n0 = blockIdx.x * 64;

  const bf16* arow0 = A + (size_t)(m0 + lane15) * Cc;
  const bf16* arow1 = A + (size_t)(m0 + 16 + lane15) * Cc;
  v8f acc[2][4] = {};

  v16bf a0 = load_a_bf16(arow0, 0, half);
  v16bf a1 = load_a_bf16(arow1, 0, half);
  v16bf bc[4];
#pragma unroll
  for (int j = 0; j < 4; ++j)
    bc[j] = load_b_bf16T(WT, Cc, n0 + j * 16 + lane15, 0, half);

#pragma unroll 2
  for (int kb = 0; kb < Cc; kb += 32) {
    const int kn = (kb + 32 < Cc) ? (kb + 32) : kb;
    __builtin_prefetch(arow0 + kn + 96, 0, 1);
    const v16bf na0 = load_a_bf16(arow0, kn, half);
    const v16bf na1 = load_a_bf16(arow1, kn, half);
    v16bf nb[4];
#pragma unroll
    for (int j = 0; j < 4; ++j)
      nb[j] = load_b_bf16T(WT, Cc, n0 + j * 16 + lane15, kn, half);
#pragma unroll
    for (int j = 0; j < 4; ++j) {
      acc[0][j] = WMMA_BF16(a0, bc[j], acc[0][j]);
      acc[1][j] = WMMA_BF16(a1, bc[j], acc[1][j]);
    }
    a0 = na0; a1 = na1;
#pragma unroll
    for (int j = 0; j < 4; ++j) bc[j] = nb[j];
  }

#pragma unroll
  for (int j = 0; j < 4; ++j) {
    const int gn = n0 + j * 16 + lane15;
#pragma unroll
    for (int s = 0; s < 2; ++s) {
#pragma unroll
      for (int r = 0; r < 8; ++r) {
        const int gm = m0 + s * 16 + r + 8 * half;
        out[(size_t)gm * Cc + gn] = acc[s][j][r];
      }
    }
  }
}

// ---------------------------------------------------------------------------
extern "C" void kernel_launch(void* const* d_in, const int* in_sizes, int n_in,
                              void* d_out, int out_size, void* d_ws, size_t ws_size,
                              hipStream_t stream) {
  (void)in_sizes; (void)n_in; (void)out_size; (void)ws_size;
  const float* x    = (const float*)d_in[0];
  const float* wqkv = (const float*)d_in[1];
  const float* wout = (const float*)d_in[2];
  float* out = (float*)d_out;

  // Workspace (bf16 elements): xbf 8388608, wqkvT 3145728, woutT 1048576,
  // q/k/v^T 3*8388608, attn 8388608  -> ~88 MB total.
  const size_t elems = (size_t)Bb * Hh * Tt * HD;    // 8388608
  bf16* xbf   = (bf16*)d_ws;
  bf16* wqkvT = xbf + elems;
  bf16* woutT = wqkvT + (size_t)Cc * Nqkv;
  bf16* qws   = woutT + (size_t)Cc * Cc;
  bf16* kws   = qws + elems;
  bf16* vws   = kws + elems;
  bf16* aws   = vws + elems;

  dim3 b256(256), b128(128);

  cvt_x_kernel<<<dim3(Mrows * Cc / (256 * 4)), b256, 0, stream>>>(x, xbf);
  transpose_cvt_kernel<<<dim3(Cc / 64, Nqkv / 64), b256, 0, stream>>>(
      wqkv, wqkvT, Cc, Nqkv);
  transpose_cvt_kernel<<<dim3(Cc / 64, Cc / 64), b256, 0, stream>>>(
      wout, woutT, Cc, Cc);

  qkv_gemm_kernel<<<dim3(Nqkv / 64, Mrows / 256), b256, 0, stream>>>(
      xbf, wqkvT, qws, kws, vws);

  flash_attn_kernel<<<dim3(Tt / 64, Bb * Hh), b128, 0, stream>>>(
      qws, kws, vws, aws);

  out_proj_kernel<<<dim3(Cc / 64, Mrows / 256), b256, 0, stream>>>(
      aws, woutT, out);
}
